// SparseMoE_8504035246115
// MI455X (gfx1250) — matrix operand order repaired and checked by reading, not compiled, not run on hardware
//
#include <hip/hip_runtime.h>
#include <hip/hip_bf16.h>

// ---------------- types for WMMA ----------------
typedef __attribute__((ext_vector_type(16))) _Float16 v16h;
typedef __attribute__((ext_vector_type(8)))  _Float16 v8h;
typedef __attribute__((ext_vector_type(8)))  float    v8f;
typedef int v4i __attribute__((vector_size(16)));   // matches async-LDS builtin param

union AFrag { v16h v; v8h h[2]; };

// ---------------- problem constants ----------------
#define NPTS  131072
#define CIN   3
#define NEXP  8
#define HID   256
#define RHID  352
#define NHIDL 3
#define W0C   30.0f

#define MT     64         // points per block tile
#define ASTR   264        // padded LDS row stride in halves (132 dwords == 4 mod 64 banks)
#define KSLICE 32         // k-rows staged per step
#define NSLICE (HID / KSLICE)

// ---------------- async global->LDS staging ----------------
#if __has_builtin(__builtin_amdgcn_global_load_async_to_lds_b128)
#define HAVE_ASYNC_LDS 1
#else
#define HAVE_ASYNC_LDS 0
#endif

__device__ __forceinline__ void wait_async_all() {
#if HAVE_ASYNC_LDS
#if __has_builtin(__builtin_amdgcn_s_wait_asynccnt)
    __builtin_amdgcn_s_wait_asynccnt(0);
#else
    asm volatile("s_wait_asynccnt 0x0" ::: "memory");
#endif
#endif
}

// stage one 32x256 f16 weight slice (16 KB) into LDS; 256 threads x 4 chunks of 16 B
__device__ __forceinline__ void stage_slice(const _Float16* __restrict__ Wg, int k0,
                                            _Float16 (* __restrict__ wb)[ASTR], int tid)
{
#pragma unroll
    for (int j = 0; j < 4; ++j) {
        int idx   = tid + j * 256;
        int row   = idx >> 5;          // 0..31
        int chunk = idx & 31;          // 0..31 (x8 halves)
        const _Float16* src = Wg + (size_t)(k0 + row) * HID + chunk * 8;
        _Float16*       dst = &wb[row][chunk * 8];
#if HAVE_ASYNC_LDS
        __builtin_amdgcn_global_load_async_to_lds_b128((v4i*)src, (v4i*)dst, 0, 0);
#else
        *(v8h*)dst = *(const v8h*)src;
#endif
    }
}

// ---------------- core GEMM: [64x256(f16 in LDS)] x [256x256(f16 global)] ----------------
// waves tiled 4(M) x 2(N); acc must be pre-initialized (bias); act rows stride ASTR.
__device__ __forceinline__ void gemm_layer(
    const _Float16 (* __restrict__ act)[ASTR],
    _Float16 (* __restrict__ wbuf)[KSLICE][ASTR],   // [2][32][264] double buffer
    const _Float16* __restrict__ Wg,                // 256x256 row-major f16
    v8f acc[8], int tid)
{
    const int lane = tid & 31, wave = tid >> 5;
    const int m_base = (wave >> 1) * 16, n_base = (wave & 1) * 128;
    const int lo = lane & 15, hi = lane >> 4;

    stage_slice(Wg, 0, wbuf[0], tid);
    for (int i = 0; i < NSLICE; ++i) {
        wait_async_all();          // this wave's staged slice i is in LDS
        __syncthreads();           // ... and everyone else's portion too
        if (i + 1 < NSLICE)        // prefetch next slice while computing this one
            stage_slice(Wg, (i + 1) * KSLICE, wbuf[(i + 1) & 1], tid);

        const _Float16 (* __restrict__ wb)[ASTR] = wbuf[i & 1];
        const int k0 = i * KSLICE;

        // A fragment: lane holds row M = m_base+lo, K split per 16-bit A layout
        AFrag a;
        const _Float16* arow = &act[m_base + lo][0];
        a.h[0] = *(const v8h*)(arow + k0 + hi * 8);
        a.h[1] = *(const v8h*)(arow + k0 + 16 + hi * 8);

        // B fragments from LDS: lane holds column N = n_base+t*16+lo, K split per 16-bit layout
        const _Float16* wk = &wb[0][n_base + lo];
#pragma unroll
        for (int t = 0; t < 8; ++t) {
            AFrag b;
#pragma unroll
            for (int s = 0; s < 8; ++s) {
                b.v[s]     = wk[(hi * 8 + s) * ASTR + t * 16];
                b.v[s + 8] = wk[(16 + hi * 8 + s) * ASTR + t * 16];
            }
            acc[t] = __builtin_amdgcn_wmma_f32_16x16x32_f16(
                false, a.v, false, b.v, (short)0, acc[t], false, false);
        }
    }
}

// ======================================================================
// 1) f32 -> f16 weight conversion (Wh, Wo, Wd1)
// ======================================================================
__global__ __launch_bounds__(256) void convert_kernel(
    const float* __restrict__ Wh, const float* __restrict__ Wo,
    const float* __restrict__ Wd1,
    _Float16* __restrict__ Whf, _Float16* __restrict__ Wof,
    _Float16* __restrict__ Wd1f)
{
    const int NH = NEXP * NHIDL * HID * HID;
    const int NO = NEXP * HID * HID;
    const int ND = HID * HID;
    int i = blockIdx.x * blockDim.x + threadIdx.x;
    if (i < NH)                 Whf[i]            = (_Float16)Wh[i];
    else if (i < NH + NO)       Wof[i - NH]       = (_Float16)Wo[i - NH];
    else if (i < NH + NO + ND)  Wd1f[i - NH - NO] = (_Float16)Wd1[i - NH - NO];
}

// ======================================================================
// 2) Router: logits = relu(ctx@Wr1+br1)@Wr2+br2, top-2 softmax, scatter
// ======================================================================
__global__ __launch_bounds__(256) void router_kernel(
    const float* __restrict__ ctx,
    const float* __restrict__ Wr1, const float* __restrict__ br1,
    const float* __restrict__ Wr2, const float* __restrict__ br2,
    int* __restrict__ counts, int* __restrict__ elist,
    float* __restrict__ egate)
{
    int p = blockIdx.x * blockDim.x + threadIdx.x;
    if (p >= NPTS) return;
    float c0 = ctx[p * 3 + 0], c1 = ctx[p * 3 + 1], c2 = ctx[p * 3 + 2];
    float logit[NEXP];
#pragma unroll
    for (int e = 0; e < NEXP; ++e) logit[e] = br2[e];
    for (int j = 0; j < RHID; ++j) {
        float a = fmaf(c0, Wr1[j],
                  fmaf(c1, Wr1[RHID + j],
                  fmaf(c2, Wr1[2 * RHID + j], br1[j])));
        a = fmaxf(a, 0.0f);
        const float* w2 = Wr2 + j * NEXP;
#pragma unroll
        for (int e = 0; e < NEXP; ++e) logit[e] = fmaf(a, w2[e], logit[e]);
    }
    int e0 = 0; float v0 = logit[0];
#pragma unroll
    for (int e = 1; e < NEXP; ++e) if (logit[e] > v0) { v0 = logit[e]; e0 = e; }
    int e1 = -1; float v1 = -3.0e38f;
#pragma unroll
    for (int e = 0; e < NEXP; ++e)
        if (e != e0 && logit[e] > v1) { v1 = logit[e]; e1 = e; }
    float w0 = 1.0f / (1.0f + __expf(v1 - v0));
    float w1 = 1.0f - w0;
    int pos0 = atomicAdd(&counts[e0], 1);
    elist[e0 * NPTS + pos0] = p; egate[e0 * NPTS + pos0] = w0;
    int pos1 = atomicAdd(&counts[e1], 1);
    elist[e1 * NPTS + pos1] = p; egate[e1 * NPTS + pos1] = w1;
}

// ======================================================================
// 3) Expert kernel: 64-point tile through full SIREN chain of one expert
// ======================================================================
__global__ __launch_bounds__(256) void expert_kernel(
    const float* __restrict__ x,
    const int*   __restrict__ elist, const float* __restrict__ egate,
    const int*   __restrict__ counts,
    const float* __restrict__ W0,  const float* __restrict__ b0,
    const _Float16* __restrict__ Whf, const float* __restrict__ bh,
    const _Float16* __restrict__ Wof, const float* __restrict__ bo,
    float* __restrict__ final_acc)
{
    const int e    = blockIdx.y;
    const int cnt  = counts[e];
    const int base = blockIdx.x * MT;
    if (base >= cnt) return;
    const int rows = min(MT, cnt - base);

    __shared__ _Float16 act[MT][ASTR];
    __shared__ _Float16 wbuf[2][KSLICE][ASTR];
    __shared__ int   spidx[MT];
    __shared__ float sgate[MT];

    const int tid = threadIdx.x;
    if (tid < MT) {
        int r = base + tid;
        spidx[tid] = (tid < rows) ? elist[e * NPTS + r] : 0;
        sgate[tid] = (tid < rows) ? egate[e * NPTS + r] : 0.0f;
    }
    __syncthreads();

    // ---- layer 0: [MT x 3] @ [3 x 256] + bias, sin(30*z), VALU path ----
    {
        const float* w = W0 + (size_t)e * CIN * HID;
        const float* b = b0 + e * HID;
        for (int i = tid; i < MT * HID; i += 256) {
            int m = i >> 8, n = i & (HID - 1);
            int p = spidx[m];
            float x0 = x[p * 3 + 0], x1 = x[p * 3 + 1], x2 = x[p * 3 + 2];
            float z = fmaf(x0, w[n],
                      fmaf(x1, w[HID + n],
                      fmaf(x2, w[2 * HID + n], b[n])));
            act[m][n] = (_Float16)__sinf(W0C * z);
        }
    }
    __syncthreads();

    const int lane = tid & 31, wave = tid >> 5;
    const int m_base = (wave >> 1) * 16, n_base = (wave & 1) * 128;
    const int lo = lane & 15, hi = lane >> 4;

    for (int layer = 0; layer < 4; ++layer) {
        const _Float16* W = (layer < NHIDL)
            ? (Whf + ((size_t)(e * NHIDL + layer) << 16))
            : (Wof + ((size_t)e << 16));
        const float* bias = (layer < NHIDL)
            ? (bh + (e * NHIDL + layer) * HID)
            : (bo + e * HID);

        v8f acc[8];
#pragma unroll
        for (int t = 0; t < 8; ++t) {
            float bv = bias[n_base + t * 16 + lo];
#pragma unroll
            for (int v = 0; v < 8; ++v) acc[t][v] = bv;
        }

        gemm_layer(act, wbuf, W, acc, tid);
        __syncthreads();   // all waves done reading act & wbuf

        if (layer < NHIDL) {
#pragma unroll
            for (int t = 0; t < 8; ++t)
#pragma unroll
                for (int v = 0; v < 8; ++v) {
                    int m = m_base + v + 8 * hi;
                    int n = n_base + t * 16 + lo;
                    act[m][n] = (_Float16)__sinf(W0C * acc[t][v]);
                }
            __syncthreads();
        } else {
#pragma unroll
            for (int t = 0; t < 8; ++t)
#pragma unroll
                for (int v = 0; v < 8; ++v) {
                    int m = m_base + v + 8 * hi;
                    if (m < rows) {
                        int n = n_base + t * 16 + lo;
                        int p = spidx[m];
                        unsafeAtomicAdd(&final_acc[(size_t)p * HID + n],
                                        sgate[m] * acc[t][v]);
                    }
                }
        }
    }
}

// ======================================================================
// 4) Decoder: out = relu(final@Wd1+bd1)@Wd2 + bd2, WMMA + LDS reduce
// ======================================================================
__global__ __launch_bounds__(256) void decoder_kernel(
    const float* __restrict__ final_acc,
    const _Float16* __restrict__ Wd1f, const float* __restrict__ bd1,
    const float* __restrict__ Wd2, const float* __restrict__ bd2,
    float* __restrict__ out)
{
    const int base = blockIdx.x * MT;
    __shared__ _Float16 act[MT][ASTR];
    __shared__ _Float16 wbuf[2][KSLICE][ASTR];
    __shared__ float red[MT];
    const int tid = threadIdx.x;

    for (int i = tid; i < MT * HID; i += 256) {
        int m = i >> 8, n = i & (HID - 1);
        act[m][n] = (_Float16)final_acc[(size_t)(base + m) * HID + n];
    }
    if (tid < MT) red[tid] = 0.0f;
    __syncthreads();

    const int lane = tid & 31, wave = tid >> 5;
    const int m_base = (wave >> 1) * 16, n_base = (wave & 1) * 128;
    const int lo = lane & 15, hi = lane >> 4;

    v8f acc[8];
#pragma unroll
    for (int t = 0; t < 8; ++t) {
        float bv = bd1[n_base + t * 16 + lo];
#pragma unroll
        for (int v = 0; v < 8; ++v) acc[t][v] = bv;
    }

    gemm_layer(act, wbuf, Wd1f, acc, tid);

    // relu then dot with Wd2 over this wave's N range; reduce via LDS
    float partial[8];
#pragma unroll
    for (int v = 0; v < 8; ++v) partial[v] = 0.0f;
#pragma unroll
    for (int t = 0; t < 8; ++t) {
        float w2 = Wd2[n_base + t * 16 + lo];
#pragma unroll
        for (int v = 0; v < 8; ++v)
            partial[v] = fmaf(fmaxf(acc[t][v], 0.0f), w2, partial[v]);
    }
#pragma unroll
    for (int v = 0; v < 8; ++v)
        atomicAdd(&red[m_base + v + 8 * hi], partial[v]);   // ds_add_f32
    __syncthreads();
    if (tid < MT) out[base + tid] = red[tid] + bd2[0];
}

// ======================================================================
// launch
// ======================================================================
extern "C" void kernel_launch(void* const* d_in, const int* in_sizes, int n_in,
                              void* d_out, int out_size, void* d_ws, size_t ws_size,
                              hipStream_t stream) {
    const float* x    = (const float*)d_in[0];
    const float* ctx  = (const float*)d_in[1];
    const float* Wr1  = (const float*)d_in[2];
    const float* br1  = (const float*)d_in[3];
    const float* Wr2  = (const float*)d_in[4];
    const float* br2  = (const float*)d_in[5];
    const float* W0   = (const float*)d_in[6];
    const float* b0   = (const float*)d_in[7];
    const float* Wh   = (const float*)d_in[8];
    const float* bh   = (const float*)d_in[9];
    const float* Wo   = (const float*)d_in[10];
    const float* bo   = (const float*)d_in[11];
    const float* Wd1  = (const float*)d_in[12];
    const float* bd1  = (const float*)d_in[13];
    const float* Wd2  = (const float*)d_in[14];
    const float* bd2  = (const float*)d_in[15];
    float* out = (float*)d_out;

    // workspace layout
    char* ws = (char*)d_ws;
    size_t off = 0;
    float* final_acc = (float*)(ws + off); off += (size_t)NPTS * HID * 4;      // 128 MB
    _Float16* Whf  = (_Float16*)(ws + off); off += (size_t)NEXP * NHIDL * HID * HID * 2;
    _Float16* Wof  = (_Float16*)(ws + off); off += (size_t)NEXP * HID * HID * 2;
    _Float16* Wd1f = (_Float16*)(ws + off); off += (size_t)HID * HID * 2;
    int* counts    = (int*)(ws + off);      off += 128;
    int* elist     = (int*)(ws + off);      off += (size_t)NEXP * NPTS * 4;
    float* egate   = (float*)(ws + off);    off += (size_t)NEXP * NPTS * 4;

    (void)hipMemsetAsync(final_acc, 0, (size_t)NPTS * HID * 4, stream);
    (void)hipMemsetAsync(counts, 0, NEXP * sizeof(int), stream);

    {   // weight conversion
        const int tot = NEXP * NHIDL * HID * HID + NEXP * HID * HID + HID * HID;
        convert_kernel<<<(tot + 255) / 256, 256, 0, stream>>>(Wh, Wo, Wd1, Whf, Wof, Wd1f);
    }
    router_kernel<<<NPTS / 256, 256, 0, stream>>>(ctx, Wr1, br1, Wr2, br2,
                                                  counts, elist, egate);
    {
        dim3 grid(NPTS / MT, NEXP);   // worst case: every point on one expert
        expert_kernel<<<grid, 256, 0, stream>>>(x, elist, egate, counts,
                                                W0, b0, Whf, bh, Wof, bo,
                                                final_acc);
    }
    decoder_kernel<<<NPTS / MT, 256, 0, stream>>>(final_acc, Wd1f, bd1,
                                                  Wd2, bd2, out);
}
